// LIIF_61443802136760
// MI455X (gfx1250) — matrix operand order repaired and checked by reading, not compiled, not run on hardware
//
#include <hip/hip_runtime.h>

typedef _Float16 half_t;
typedef __attribute__((ext_vector_type(16))) _Float16 v16h;
typedef __attribute__((ext_vector_type(8)))  _Float16 v8h;
typedef __attribute__((ext_vector_type(8)))  float    v8f;

#define SW 392            // LDS row stride in halfs (384 + 8 pad -> stride % 64 dwords == 4)
#define ROWS 32           // rows per workgroup (one wave, two 16-row M tiles)
#define EPS_F 1e-3f

// packed-weight layout offsets (halfs) inside workspace
#define OFF_WINIT 0
#define OFF_W0    32768
#define OFF_W1    131072
#define OFF_W2    196608
#define OFF_W3    262144
#define OFF_W4    360448
#define OFF_GRID  425984   // repacked grid: [y][x][c] f16, 16*16*128

union AFrag { v16h v; v8h h[2]; };

// 32-row stripe through one layer: out[32x256] = sin(A[32xK] @ Wpacked + bias).
// A/out in LDS (row stride SW halfs). B fragments shared across both M tiles;
// A fragments loaded once per kc per n-block (16 accumulating output tiles live).
template<int K>
__device__ __forceinline__ void layer_gemm(const half_t* lA, half_t* lO,
                                           const half_t* __restrict__ wp,
                                           const float* __restrict__ bias, int lane)
{
    const int m  = lane & 15;
    const int hi = lane >> 4;
    const half_t* arow0 = lA + m * SW;          // M tile 0: rows 0..15
    const half_t* arow1 = lA + (16 + m) * SW;   // M tile 1: rows 16..31
    #pragma unroll
    for (int nb = 0; nb < 2; ++nb) {
        v8f acc[2][8];
        #pragma unroll
        for (int mt = 0; mt < 2; ++mt)
            #pragma unroll
            for (int t = 0; t < 8; ++t) acc[mt][t] = (v8f){};
        for (int kc = 0; kc < K; kc += 32) {
            // A fragments per ISA 16-bit A 16x32 layout
            AFrag a0, a1;
            a0.h[0] = *(const v8h*)(arow0 + kc + 8 * hi);
            a0.h[1] = *(const v8h*)(arow0 + kc + 16 + 8 * hi);
            a1.h[0] = *(const v8h*)(arow1 + kc + 8 * hi);
            a1.h[1] = *(const v8h*)(arow1 + kc + 16 + 8 * hi);
            const half_t* wpt = wp + (size_t)(((nb * 8) * K + kc + lane) << 4);
            #pragma unroll
            for (int t = 0; t < 8; ++t) {
                v16h b = *(const v16h*)(wpt + (size_t)t * (K << 4));
                acc[0][t] = __builtin_amdgcn_wmma_f32_16x16x32_f16(
                    false, a0.v, false, b, (short)0, acc[0][t], false, false);
                acc[1][t] = __builtin_amdgcn_wmma_f32_16x16x32_f16(
                    false, a1.v, false, b, (short)0, acc[1][t], false, false);
            }
        }
        // D layout: VGPR r -> row r + 8*hi (+16*mt); col = 16*(nb*8+t) + m
        #pragma unroll
        for (int t = 0; t < 8; ++t) {
            const int col = ((nb * 8 + t) << 4) + m;
            const float bv = bias[col];
            #pragma unroll
            for (int mt = 0; mt < 2; ++mt) {
                half_t* ocol = lO + (mt * 16 + (hi << 3)) * SW + col;
                #pragma unroll
                for (int r = 0; r < 8; ++r)
                    ocol[r * SW] = (half_t)__sinf(acc[mt][t][r] + bv);
            }
        }
    }
}

extern "C" __global__ void pack_w_kernel(const float* __restrict__ src,
                                         half_t* __restrict__ dst, int K, int total)
{
    int i = blockIdx.x * blockDim.x + threadIdx.x;
    if (i >= total) return;
    // dst[(t*K + kc + l)*16 + j] = w[(t*16 + l%16)*K + kc + (j<8 ? 8*(l/16)+j : 16+8*(l/16)+j-8)]
    int j  = i & 15;
    int tk = i >> 4;
    int kk = tk % K;
    int t  = tk / K;
    int kl = kk & 31;
    int k  = (kk - kl) + ((j < 8) ? (8 * (kl >> 4) + j) : (16 + 8 * (kl >> 4) + j - 8));
    dst[i] = (half_t)src[(t * 16 + (kl & 15)) * K + k];
}

extern "C" __global__ void pack_grid_kernel(const float* __restrict__ grid,
                                            half_t* __restrict__ dst)
{
    // grid: [1,128,16,16] f32 -> dst[(y*16+x)*128 + c] f16
    int i = blockIdx.x * blockDim.x + threadIdx.x;   // 32768 total
    if (i >= 128 * 256) return;
    int c  = i & 127;
    int yx = i >> 7;
    dst[i] = (half_t)grid[c * 256 + yx];
}

extern "C" __global__ __launch_bounds__(32)
void liif_main(const float* __restrict__ x, const half_t* __restrict__ wpk,
               const float* __restrict__ b_init, const float* __restrict__ b0,
               const float* __restrict__ b1, const float* __restrict__ b2,
               const float* __restrict__ b3, const float* __restrict__ b4,
               const float* __restrict__ w_out, const float* __restrict__ b_out,
               float* __restrict__ out, int Ntot)
{
    __shared__ half_t smem[2 * ROWS * SW];
    const int lane = threadIdx.x;             // single wave: 0..31
    half_t* buf0 = smem;
    half_t* buf1 = smem + ROWS * SW;
    const int rowBase = blockIdx.x * ROWS;
    const half_t* gpk = wpk + OFF_GRID;

    // ---- Stage 0: bilinear grid-sample (border, align_corners=False); lane -> row lane
    {
        int r = rowBase + lane; if (r > Ntot - 1) r = Ntot - 1;
        float xc = x[2 * r + 0], yc = x[2 * r + 1];
        float ix = (xc + 1.0f) * 8.0f - 0.5f;   // ((xc+1)*16 - 1) * 0.5
        float iy = (yc + 1.0f) * 8.0f - 0.5f;
        float x0f = floorf(ix), y0f = floorf(iy);
        float wx = ix - x0f, wy = iy - y0f;
        int x0 = (int)x0f, y0 = (int)y0f;
        int x1 = x0 + 1,   y1 = y0 + 1;
        x0 = min(max(x0, 0), 15); x1 = min(max(x1, 0), 15);
        y0 = min(max(y0, 0), 15); y1 = min(max(y1, 0), 15);
        float w00 = (1.0f - wx) * (1.0f - wy), w01 = wx * (1.0f - wy);
        float w10 = (1.0f - wx) * wy,          w11 = wx * wy;
        const half_t* g00 = gpk + (y0 * 16 + x0) * 128;
        const half_t* g01 = gpk + (y0 * 16 + x1) * 128;
        const half_t* g10 = gpk + (y1 * 16 + x0) * 128;
        const half_t* g11 = gpk + (y1 * 16 + x1) * 128;
        half_t* pr  = buf0 + lane * SW + 256;   // raw p (init-GEMM A operand)
        half_t* spr = buf1 + lane * SW + 256;   // sin(p) for the skip-concat layers
        #pragma unroll 2
        for (int cb = 0; cb < 128; cb += 8) {
            v8h v00 = *(const v8h*)(g00 + cb);
            v8h v01 = *(const v8h*)(g01 + cb);
            v8h v10 = *(const v8h*)(g10 + cb);
            v8h v11 = *(const v8h*)(g11 + cb);
            #pragma unroll
            for (int j = 0; j < 8; ++j) {
                float p = (float)v00[j] * w00 + (float)v01[j] * w01
                        + (float)v10[j] * w10 + (float)v11[j] * w11;
                pr[cb + j]  = (half_t)p;
                spr[cb + j] = (half_t)__sinf(p);
            }
        }
    }
    // single wave: LDS ops are in program order, no barriers needed

    // ---- init: buf0[:,0:256] = sin(p @ Winit^T + b_init), A = raw p at buf0[:,256:384]
    layer_gemm<128>(buf0 + 256, buf0, wpk + OFF_WINIT, b_init, lane);

    // replace raw p with sin(p) so buf0 = [sin(h0) | sin(p)] for layer 0
    {
        const v8h* s = (const v8h*)(buf1 + lane * SW + 256);
        v8h*       d = (v8h*)(buf0 + lane * SW + 256);
        #pragma unroll
        for (int i = 0; i < 16; ++i) d[i] = s[i];
    }

    layer_gemm<384>(buf0, buf1, wpk + OFF_W0, b0, lane);   // L0 (concat)
    layer_gemm<256>(buf1, buf0, wpk + OFF_W1, b1, lane);   // L1
    layer_gemm<256>(buf0, buf1, wpk + OFF_W2, b2, lane);   // L2
    layer_gemm<384>(buf1, buf0, wpk + OFF_W3, b3, lane);   // L3 (sp still in buf1[:,256:384])
    layer_gemm<256>(buf0, buf1, wpk + OFF_W4, b4, lane);   // L4

    // ---- final 256 -> 1 dot + fat sigmoid; lane -> row lane (w_out loads are uniform)
    {
        const half_t* hr = buf1 + lane * SW;
        float s = 0.0f;
        #pragma unroll 8
        for (int k = 0; k < 256; ++k)
            s += (float)hr[k] * w_out[k];
        int r = rowBase + lane;
        if (r < Ntot) {
            float t  = s + b_out[0];
            float sg = 1.0f / (1.0f + __expf(-t));
            out[r] = sg * (1.0f + 2.0f * EPS_F) - EPS_F;
        }
    }
}

extern "C" void kernel_launch(void* const* d_in, const int* in_sizes, int n_in,
                              void* d_out, int out_size, void* d_ws, size_t ws_size,
                              hipStream_t stream)
{
    const float* x      = (const float*)d_in[0];
    const float* grid   = (const float*)d_in[1];
    const float* w_init = (const float*)d_in[2];
    const float* b_init = (const float*)d_in[3];
    const float* w_out  = (const float*)d_in[4];
    const float* b_out  = (const float*)d_in[5];
    const float* w[5];
    const float* b[5];
    for (int i = 0; i < 5; ++i) {
        w[i] = (const float*)d_in[6 + 2 * i];
        b[i] = (const float*)d_in[7 + 2 * i];
    }
    float*  out = (float*)d_out;
    half_t* wpk = (half_t*)d_ws;

    // Pack weights into WMMA B-fragment order (f32 -> f16) and the grid into [y][x][c] f16.
    const int    Ks[6]   = {128, 384, 256, 256, 384, 256};
    const size_t offs[6] = {OFF_WINIT, OFF_W0, OFF_W1, OFF_W2, OFF_W3, OFF_W4};
    const float* srcs[6] = {w_init, w[0], w[1], w[2], w[3], w[4]};
    for (int l = 0; l < 6; ++l) {
        int total = 256 * Ks[l];
        pack_w_kernel<<<(total + 255) / 256, 256, 0, stream>>>(srcs[l], wpk + offs[l], Ks[l], total);
    }
    pack_grid_kernel<<<(128 * 256 + 255) / 256, 256, 0, stream>>>(grid, wpk + OFF_GRID);

    int Ntot   = in_sizes[0] / 2;                 // B*H*W points
    int blocks = (Ntot + ROWS - 1) / ROWS;
    liif_main<<<blocks, 32, 0, stream>>>(
        x, wpk, b_init, b[0], b[1], b[2], b[3], b[4], w_out, b_out, out, Ntot);
}
